// KTM_71339406786898
// MI455X (gfx1250) — matrix operand order repaired and checked by reading, not compiled, not run on hardware
//
#include <hip/hip_runtime.h>
#include <hip/hip_bf16.h>

typedef __attribute__((ext_vector_type(16))) _Float16 v16h;
typedef __attribute__((ext_vector_type(8)))  _Float16 v8h;
typedef __attribute__((ext_vector_type(8)))  float    v8f;

#define BB 8
#define CC 32
#define CQ 16
#define HH 64
#define WW 64
#define HWP 4096   // H*W
#define WAVES 8

// ---- DPP butterfly reductions (VALU only; no LDS round trips) --------------
template <int CTRL>
__device__ __forceinline__ float dpp_fmax(float x) {
    int yi = __builtin_amdgcn_update_dpp(0, __float_as_int(x), CTRL, 0xF, 0xF, true);
    return fmaxf(x, __int_as_float(yi));
}
template <int CTRL>
__device__ __forceinline__ float dpp_fadd(float x) {
    int yi = __builtin_amdgcn_update_dpp(0, __float_as_int(x), CTRL, 0xF, 0xF, true);
    return x + __int_as_float(yi);
}
// max over all 32 lanes (broadcast): 4 DPP steps within 16-lane rows + 1 cross-half
__device__ __forceinline__ float wave_max32(float x) {
    x = dpp_fmax<0xB1>(x);   // quad_perm [1,0,3,2]  (xor 1)
    x = dpp_fmax<0x4E>(x);   // quad_perm [2,3,0,1]  (xor 2)
    x = dpp_fmax<0x141>(x);  // row_half_mirror      (xor 7 within 8)
    x = dpp_fmax<0x140>(x);  // row_mirror           (xor 15 within 16)
    x = fmaxf(x, __shfl_xor(x, 16));
    return x;
}
// sum within each 16-lane half (broadcast within the half): DPP only
__device__ __forceinline__ float half_sum16(float x) {
    x = dpp_fadd<0xB1>(x);
    x = dpp_fadd<0x4E>(x);
    x = dpp_fadd<0x141>(x);
    x = dpp_fadd<0x140>(x);
    return x;
}

// interleaved j-position within each 32-block (shared by P staging & V storage)
__device__ __forceinline__ int jpos(int w) {           // w in [0,32)
    return (w < 16) ? (2 * w) : (2 * (w - 16) + 1);
}

// ---------------------------------------------------------------------------
// Kernel 1: 1x1 projections q,k (16ch) and v2,v3 (32ch), stored as f16.
//   qh layout: [b][pixel][16]      (A-fragment friendly, contiguous per pixel)
//   kh layout: [b][pixel][32]      (B-fragment friendly; ch 16..31 zeroed so
//                                   lanes 16-31 load the K-padding from memory)
//   v2h/v3h layout: [b][c][4096], with the j-order inside every 32-block
//   interleaved (jpos) to match the packed P staging order.
// ---------------------------------------------------------------------------
__global__ __launch_bounds__(256) void proj_kernel(
    const float* __restrict__ x2, const float* __restrict__ x3,
    const float* __restrict__ wq, const float* __restrict__ bq,
    const float* __restrict__ wk, const float* __restrict__ bk,
    const float* __restrict__ wv2, const float* __restrict__ bv2,
    const float* __restrict__ wv3, const float* __restrict__ bv3,
    _Float16* __restrict__ qh, _Float16* __restrict__ kh,
    _Float16* __restrict__ v2h, _Float16* __restrict__ v3h)
{
    __shared__ float swq[CQ * CC], swk[CQ * CC];
    __shared__ float swv2[CC * CC], swv3[CC * CC];
    for (int t = threadIdx.x; t < CQ * CC; t += 256) { swq[t] = wq[t]; swk[t] = wk[t]; }
    for (int t = threadIdx.x; t < CC * CC; t += 256) { swv2[t] = wv2[t]; swv3[t] = wv3[t]; }
    __syncthreads();

    int p = blockIdx.x * 256 + threadIdx.x;
    if (p >= BB * HWP) return;
    int b = p >> 12;
    int i = p & (HWP - 1);

    float qa[CQ], ka[CQ], v2a[CC], v3a[CC];
#pragma unroll
    for (int co = 0; co < CQ; ++co) { qa[co] = bq[co]; ka[co] = bk[co]; }
#pragma unroll
    for (int co = 0; co < CC; ++co) { v2a[co] = bv2[co]; v3a[co] = bv3[co]; }

    for (int ci = 0; ci < CC; ++ci) {
        float u = x2[((size_t)(b * CC + ci)) * HWP + i];
        float v = x3[((size_t)(b * CC + ci)) * HWP + i];
        float s = u + v;
        float m = u * v;
#pragma unroll
        for (int co = 0; co < CQ; ++co) {
            qa[co] += swq[co * CC + ci] * s;
            ka[co] += swk[co * CC + ci] * m;
        }
#pragma unroll
        for (int co = 0; co < CC; ++co) {
            v2a[co] += swv2[co * CC + ci] * u;
            v3a[co] += swv3[co * CC + ci] * v;
        }
    }
#pragma unroll
    for (int co = 0; co < CQ; ++co) {
        qh[(size_t)p * CQ + co] = (_Float16)qa[co];
        kh[(size_t)p * CC + co] = (_Float16)ka[co];
        kh[(size_t)p * CC + CQ + co] = (_Float16)0.0f;   // K-dim zero padding
    }
    int ji = (i & ~31) | jpos(i & 31);   // interleaved j index for V storage
#pragma unroll
    for (int co = 0; co < CC; ++co) {
        v2h[((size_t)(b * CC + co)) * HWP + ji] = (_Float16)v2a[co];
        v3h[((size_t)(b * CC + co)) * HWP + ji] = (_Float16)v3a[co];
    }
}

// ---------------------------------------------------------------------------
// Kernel 2: fused flash attention. One wave per 16-row i-tile.
//   S(16x32 cols) = Q(16x16,pad32) * K via WMMA, online softmax with a
//   TILE-WIDE running max (rescale is wave-uniform and usually skipped),
//   per-lane deferred row sums, O += P * V^T via WMMA.
//   P staged packed (cvt_pk + ds_store_b32); V is stored with matching
//   interleaved j-order, so the PV K-dimension ordering is consistent.
// ---------------------------------------------------------------------------
__global__ __launch_bounds__(256) void attn_kernel(
    const _Float16* __restrict__ qh, const _Float16* __restrict__ kh,
    const _Float16* __restrict__ v2h, const _Float16* __restrict__ v3h,
    const float* __restrict__ x2, const float* __restrict__ x3,
    const float* __restrict__ gamma2, const float* __restrict__ gamma3,
    float* __restrict__ t2, float* __restrict__ t3)
{
    __shared__ __align__(16) _Float16 pstage[WAVES][16 * 32];

    const int wave = threadIdx.x >> 5;
    const int lane = threadIdx.x & 31;
    const int half = lane >> 4;      // 0: lanes 0-15, 1: lanes 16-31
    const int l16  = lane & 15;

    const int b    = blockIdx.x >> 5;                       // 8 batches
    const int tile = (blockIdx.x & 31) * WAVES + wave;      // 0..255
    const int i0   = tile * 16;

    // ---- Q fragment (A layout, 16x32 f16; K-dims 16..31 are zero padding)
    v16h qa = {};
    {
        const _Float16* qp = qh + ((size_t)b * HWP + i0 + l16) * CQ;
        if (!half) qa = *(const v16h*)qp;
    }

    float mrun = -1e30f;     // tile-wide running max (wave-uniform)
    float l[8];              // per-lane PARTIAL row sums (reduced at the end)
#pragma unroll
    for (int r = 0; r < 8; ++r) l[r] = 0.0f;
    v8f o20 = {}, o21 = {}, o30 = {}, o31 = {};
    const v8f zc = {};

    // All lanes load real data: lanes>=16 fetch the zeroed channels 16..31.
    const _Float16* kbase  = kh + ((size_t)b * HWP + l16) * CC + half * CQ;
    const _Float16* v2base = v2h + (size_t)b * CC * HWP;
    const _Float16* v3base = v3h + (size_t)b * CC * HWP;

    for (int step = 0; step < HWP / 32; ++step) {
        const int j0 = step * 32;

        // ---- K fragments (B layout, 32x16): one contiguous 32B load per lane
        v16h kb0 = *(const v16h*)(kbase + (size_t)j0 * CC);
        v16h kb1 = *(const v16h*)(kbase + (size_t)(j0 + 16) * CC);

        v8f s0 = __builtin_amdgcn_wmma_f32_16x16x32_f16(false, qa, false, kb0,
                                                        (short)0, zc, false, false);
        v8f s1 = __builtin_amdgcn_wmma_f32_16x16x32_f16(false, qa, false, kb1,
                                                        (short)0, zc, false, false);

        // ---- tile-wide chunk max: in-lane over 16 S elements, then DPP butterfly
        float cm = s0[0];
#pragma unroll
        for (int r = 1; r < 8; ++r) cm = fmaxf(cm, s0[r]);
#pragma unroll
        for (int r = 0; r < 8; ++r) cm = fmaxf(cm, s1[r]);
        cm = wave_max32(cm);

        // ---- uniform rescale, only when the running max grows
        if (cm > mrun) {
            float a = __expf(mrun - cm);
            mrun = cm;
#pragma unroll
            for (int r = 0; r < 8; ++r) {
                l[r] *= a;
                o20[r] *= a; o21[r] *= a;
                o30[r] *= a; o31[r] *= a;
            }
        }

        // ---- P = exp(S - m): partial row sums + packed f16 staging (1 b32/row)
#pragma unroll
        for (int r = 0; r < 8; ++r) {
            float p0 = __expf(s0[r] - mrun);
            float p1 = __expf(s1[r] - mrun);
            l[r] += p0 + p1;
            int row = half ? (r + 8) : r;
            auto pk = __builtin_amdgcn_cvt_pkrtz(p0, p1);  // v_cvt_pk_rtz_f16_f32
            *reinterpret_cast<decltype(pk)*>(&pstage[wave][row * 32 + 2 * l16]) = pk;
        }
        asm volatile("" ::: "memory");  // keep P store -> load ordered (same wave, LDS in-order)

        // ---- reload P as an A fragment (D-layout -> A-layout transpose via LDS)
        v16h pa;
        {
            const _Float16* pp = &pstage[wave][l16 * 32 + half * 16];
            v8h plo = *(const v8h*)pp;
            v8h phi = *(const v8h*)(pp + 8);
#pragma unroll
            for (int t = 0; t < 8; ++t) { pa[t] = plo[t]; pa[t + 8] = phi[t]; }
        }

        // ---- V fragments (B layout, 32(j) x 16(c)); frag0 = c 0..15, frag1 = c 16..31
        v16h v2b0 = *(const v16h*)(v2base + (size_t)l16 * HWP + j0 + half * 16);
        v16h v2b1 = *(const v16h*)(v2base + (size_t)(16 + l16) * HWP + j0 + half * 16);
        v16h v3b0 = *(const v16h*)(v3base + (size_t)l16 * HWP + j0 + half * 16);
        v16h v3b1 = *(const v16h*)(v3base + (size_t)(16 + l16) * HWP + j0 + half * 16);

        o20 = __builtin_amdgcn_wmma_f32_16x16x32_f16(false, pa, false, v2b0, (short)0, o20, false, false);
        o21 = __builtin_amdgcn_wmma_f32_16x16x32_f16(false, pa, false, v2b1, (short)0, o21, false, false);
        o30 = __builtin_amdgcn_wmma_f32_16x16x32_f16(false, pa, false, v3b0, (short)0, o30, false, false);
        o31 = __builtin_amdgcn_wmma_f32_16x16x32_f16(false, pa, false, v3b1, (short)0, o31, false, false);
    }

    // ---- finalize row sums: one DPP reduction per row (within halves)
#pragma unroll
    for (int r = 0; r < 8; ++r) l[r] = half_sum16(l[r]);

    // ---- epilogue: t = gamma * (O / l) + x
    const float g2 = gamma2[0];
    const float g3 = gamma3[0];
#pragma unroll
    for (int r = 0; r < 8; ++r) {
        int i = i0 + (half ? (r + 8) : r);
        float invl = 1.0f / l[r];
        size_t idx0 = ((size_t)(b * CC + l16)) * HWP + i;
        size_t idx1 = ((size_t)(b * CC + 16 + l16)) * HWP + i;
        t2[idx0] = g2 * o20[r] * invl + x2[idx0];
        t2[idx1] = g2 * o21[r] * invl + x2[idx1];
        t3[idx0] = g3 * o30[r] * invl + x3[idx0];
        t3[idx1] = g3 * o31[r] * invl + x3[idx1];
    }
}

// ---------------------------------------------------------------------------
// Kernel 3: conv3x3 + BN + ReLU (both branches) + two 1x1s + sum + final 1x1.
// One thread per (b, pixel); weight indices are wave-uniform -> scalar loads.
// ---------------------------------------------------------------------------
__global__ __launch_bounds__(256) void tail_kernel(
    const float* __restrict__ t2, const float* __restrict__ t3,
    const float* __restrict__ w2_3, const float* __restrict__ bn2_s, const float* __restrict__ bn2_b,
    const float* __restrict__ w2_1, const float* __restrict__ b2_1,
    const float* __restrict__ w3_3, const float* __restrict__ bn3_s, const float* __restrict__ bn3_b,
    const float* __restrict__ w3_1, const float* __restrict__ b3_1,
    const float* __restrict__ wo, const float* __restrict__ bo,
    float* __restrict__ out)
{
    int p = blockIdx.x * 256 + threadIdx.x;
    if (p >= BB * HWP) return;
    int b = p >> 12;
    int pix = p & (HWP - 1);
    int y = pix >> 6;
    int x = pix & (WW - 1);

    float acc2[CC], acc3[CC];
#pragma unroll
    for (int co = 0; co < CC; ++co) { acc2[co] = 0.0f; acc3[co] = 0.0f; }

    for (int ci = 0; ci < CC; ++ci) {
        const float* b2 = t2 + ((size_t)(b * CC + ci)) * HWP;
        const float* b3 = t3 + ((size_t)(b * CC + ci)) * HWP;
#pragma unroll
        for (int dy = 0; dy < 3; ++dy) {
            int yy = y + dy - 1;
            if (yy < 0 || yy >= HH) continue;
#pragma unroll
            for (int dx = 0; dx < 3; ++dx) {
                int xx = x + dx - 1;
                if (xx < 0 || xx >= WW) continue;
                float a2 = b2[yy * WW + xx];
                float a3 = b3[yy * WW + xx];
                int wbase = ci * 9 + dy * 3 + dx;
#pragma unroll
                for (int co = 0; co < CC; ++co) {
                    acc2[co] += a2 * w2_3[co * (CC * 9) + wbase];
                    acc3[co] += a3 * w3_3[co * (CC * 9) + wbase];
                }
            }
        }
    }

    // BN + ReLU in place
#pragma unroll
    for (int co = 0; co < CC; ++co) {
        acc2[co] = fmaxf(acc2[co] * bn2_s[co] + bn2_b[co], 0.0f);
        acc3[co] = fmaxf(acc3[co] * bn3_s[co] + bn3_b[co], 0.0f);
    }

    // branch 1x1s and sum
    float h[CC];
#pragma unroll
    for (int co = 0; co < CC; ++co) {
        float s2 = b2_1[co];
        float s3 = b3_1[co];
#pragma unroll
        for (int ci = 0; ci < CC; ++ci) {
            s2 += w2_1[co * CC + ci] * acc2[ci];
            s3 += w3_1[co * CC + ci] * acc3[ci];
        }
        h[co] = s2 + s3;
    }

    // final 1x1
#pragma unroll
    for (int co = 0; co < CC; ++co) {
        float o = bo[co];
#pragma unroll
        for (int ci = 0; ci < CC; ++ci) o += wo[co * CC + ci] * h[ci];
        out[((size_t)(b * CC + co)) * HWP + pix] = o;
    }
}

// ---------------------------------------------------------------------------
extern "C" void kernel_launch(void* const* d_in, const int* in_sizes, int n_in,
                              void* d_out, int out_size, void* d_ws, size_t ws_size,
                              hipStream_t stream) {
    const float* x2     = (const float*)d_in[0];
    const float* x3     = (const float*)d_in[1];
    const float* wq     = (const float*)d_in[2];
    const float* bq     = (const float*)d_in[3];
    const float* wk     = (const float*)d_in[4];
    const float* bk     = (const float*)d_in[5];
    const float* wv2    = (const float*)d_in[6];
    const float* bv2    = (const float*)d_in[7];
    const float* wv3    = (const float*)d_in[8];
    const float* bv3    = (const float*)d_in[9];
    const float* gamma2 = (const float*)d_in[10];
    const float* gamma3 = (const float*)d_in[11];
    const float* w2_3   = (const float*)d_in[12];
    const float* bn2_s  = (const float*)d_in[13];
    const float* bn2_b  = (const float*)d_in[14];
    const float* w2_1   = (const float*)d_in[15];
    const float* b2_1   = (const float*)d_in[16];
    const float* w3_3   = (const float*)d_in[17];
    const float* bn3_s  = (const float*)d_in[18];
    const float* bn3_b  = (const float*)d_in[19];
    const float* w3_1   = (const float*)d_in[20];
    const float* b3_1   = (const float*)d_in[21];
    const float* wo     = (const float*)d_in[22];
    const float* bo     = (const float*)d_in[23];
    float* out = (float*)d_out;

    char* ws = (char*)d_ws;
    const size_t MB = 1u << 20;
    _Float16* qh  = (_Float16*)(ws + 0 * MB);   // 1 MB
    _Float16* kh  = (_Float16*)(ws + 1 * MB);   // 2 MB (32ch, upper half zeros)
    _Float16* v2h = (_Float16*)(ws + 3 * MB);   // 2 MB (interleaved j-order)
    _Float16* v3h = (_Float16*)(ws + 5 * MB);   // 2 MB (interleaved j-order)
    float*    t2  = (float*)   (ws + 7 * MB);   // 4 MB
    float*    t3  = (float*)   (ws + 11 * MB);  // 4 MB (15 MB total)

    proj_kernel<<<(BB * HWP) / 256, 256, 0, stream>>>(
        x2, x3, wq, bq, wk, bk, wv2, bv2, wv3, bv3, qh, kh, v2h, v3h);

    attn_kernel<<<BB * (HWP / (16 * WAVES)), 256, 0, stream>>>(
        qh, kh, v2h, v3h, x2, x3, gamma2, gamma3, t2, t3);

    tail_kernel<<<(BB * HWP) / 256, 256, 0, stream>>>(
        t2, t3, w2_3, bn2_s, bn2_b, w2_1, b2_1,
        w3_3, bn3_s, bn3_b, w3_1, b3_1, wo, bo, out);
}
